// SSDTarget_81415400063573
// MI455X (gfx1250) — compile-verified, matrix operand order checked by reading, not run on hardware
//
#include <hip/hip_runtime.h>
#include <stdint.h>

// ---------------- problem constants (from reference) ----------------
#define M_PRIORS 42000      // 5*(80^2+40^2+20^2)
#define BATCH    8
#define T_PER    20         // targets per image
#define NT       160
#define NCLS     80
#define IOU_TH   0.5f
#define PB       ((M_PRIORS + 255) / 256)   // 165 blocks of 256 over priors

typedef __attribute__((ext_vector_type(2)))  float     v2f;
typedef __attribute__((ext_vector_type(8)))  float     v8f;
typedef __attribute__((ext_vector_type(16))) _Float16  v16h;

// ---------------- wave32 sum-reduction via the matrix unit ----------------
// A (16x4 f32) holds the 32 lane values (vgpr0 = v, vgpr1 = 0 -> A[m][0]=v[m],
// A[m][2]=v[m+16]); B = ones => D[m][*] = v[m]+v[m+16]. Lane L's 8 C/D vgprs
// hold row-sums m = (L<16 ? 0..7 : 8..15); add them + shfl_xor(16) = full sum.
__device__ __forceinline__ float wave_reduce_sum_wmma(float v) {
  v8f c = {};
#if __has_builtin(__builtin_amdgcn_wmma_f32_16x16x4_f32)
  v2f a; a[0] = v;    a[1] = 0.0f;
  v2f b; b[0] = 1.0f; b[1] = 1.0f;
  c = __builtin_amdgcn_wmma_f32_16x16x4_f32(false, a, false, b, (short)0, c, false, false);
#else
  v16h a = {}; a[0] = (_Float16)v;          // K=0 (lanes<16) / K=8 (lanes>=16)
  v16h b;
#pragma unroll
  for (int k = 0; k < 16; ++k) b[k] = (_Float16)1.0f;
  c = __builtin_amdgcn_wmma_f32_16x16x32_f16(false, a, false, b, (short)0, c, false, false);
#endif
  float s = c[0] + c[1] + c[2] + c[3] + c[4] + c[5] + c[6] + c[7];
  s += __shfl_xor(s, 16, 32);
  return s;
}

// ---------------- helpers ----------------
__device__ __forceinline__ float iou_cxcywh(float acx, float acy, float aw, float ah,
                                            float bcx, float bcy, float bw, float bh) {
  float ax0 = acx - aw * 0.5f, ay0 = acy - ah * 0.5f;
  float ax1 = acx + aw * 0.5f, ay1 = acy + ah * 0.5f;
  float bx0 = bcx - bw * 0.5f, by0 = bcy - bh * 0.5f;
  float bx1 = bcx + bw * 0.5f, by1 = bcy + bh * 0.5f;
  float w = fmaxf(fminf(ax1, bx1) - fmaxf(ax0, bx0), 0.0f);
  float h = fmaxf(fminf(ay1, by1) - fmaxf(ay0, by0), 0.0f);
  float inter = w * h;
  return inter / (aw * ah + bw * bh - inter + 1e-9f);
}
__device__ __forceinline__ float smooth_l1(float x) {
  float ax = fabsf(x);
  return ax < 1.0f ? 0.5f * ax * ax : ax - 0.5f;
}

// ---------------- K0: zero scalar accumulators ----------------
__global__ void k_init(float* scal) {
  if (threadIdx.x < 8) scal[threadIdx.x] = 0.0f;
}

// ---------------- K1a: per (image, prior) best target (masked argmax axis 0) ----
__global__ __launch_bounds__(256) void k_best_per_prior(
    const float* __restrict__ targets, const float* __restrict__ priors,
    float* __restrict__ bo, int* __restrict__ bi) {
  int i = blockIdx.y;
  int p = blockIdx.x * 256 + threadIdx.x;
  __shared__ float tg[T_PER * 6];
  if (threadIdx.x < T_PER * 6) tg[threadIdx.x] = targets[i * T_PER * 6 + threadIdx.x];
  __syncthreads();
  if (p >= M_PRIORS) return;
  float4 pr = ((const float4*)priors)[p];
  float best = -1.0f; int bj = 0;
  for (int j = 0; j < T_PER; ++j) {            // strict > keeps first max (argmax tie-break)
    float v = iou_cxcywh(tg[j*6+2], tg[j*6+3], tg[j*6+4], tg[j*6+5], pr.x, pr.y, pr.z, pr.w);
    if (v > best) { best = v; bj = j; }
  }
  bo[i * M_PRIORS + p] = best;
  bi[i * M_PRIORS + p] = i * T_PER + bj;
}

// ---------------- K1b: per target best prior (argmax axis 1) ----------------
__global__ __launch_bounds__(256) void k_best_per_target(
    const float* __restrict__ targets, const float* __restrict__ priors,
    int* __restrict__ bp) {
  int t = blockIdx.x;
  float tcx = targets[t*6+2], tcy = targets[t*6+3], tw = targets[t*6+4], th = targets[t*6+5];
  float best = -1.0f; int bidx = 0;
  for (int p = threadIdx.x; p < M_PRIORS; p += 256) {  // ascending p: > keeps smallest index
    float4 pr = ((const float4*)priors)[p];
    float v = iou_cxcywh(tcx, tcy, tw, th, pr.x, pr.y, pr.z, pr.w);
    if (v > best) { best = v; bidx = p; }
  }
  __shared__ float sv[256];
  __shared__ int   si[256];
  sv[threadIdx.x] = best; si[threadIdx.x] = bidx;
  __syncthreads();
  for (int s = 128; s > 0; s >>= 1) {
    if (threadIdx.x < s) {
      float ov = sv[threadIdx.x + s]; int oi = si[threadIdx.x + s];
      if (ov > sv[threadIdx.x] || (ov == sv[threadIdx.x] && oi < si[threadIdx.x])) {
        sv[threadIdx.x] = ov; si[threadIdx.x] = oi;
      }
    }
    __syncthreads();
  }
  if (threadIdx.x == 0) bp[t] = si[0];
}

// ---------------- K1c: scatter local_j into best-overlap (sequential => last wins) ----
__global__ void k_scatter(const int* __restrict__ bp, float* __restrict__ bo) {
  int i = threadIdx.x;
  if (i < BATCH) {
    for (int j = 0; j < T_PER; ++j) {
      int p = bp[i * T_PER + j];
      bo[i * M_PRIORS + p] = (float)j;   // note: j==0 -> 0.0 < IOU_TH (reference quirk kept)
    }
  }
}

// ---------------- K2: single streaming pass over all 114 MB of activations ------
__global__ __launch_bounds__(256) void k_main(
    const float* __restrict__ f0, const float* __restrict__ f1, const float* __restrict__ f2,
    const float* __restrict__ targets, const float* __restrict__ priors,
    const float* __restrict__ bo, const int* __restrict__ bi,
    float* __restrict__ lossc, float* __restrict__ bcea, float* __restrict__ pfa,
    float* __restrict__ scal) {
  int i = blockIdx.y;
  int p = blockIdx.x * 256 + threadIdx.x;
  bool act = p < M_PRIORS;
  int pp = act ? p : 0;

  // decompose prior index -> (level, anchor, spatial); lanes stay coalesced in hw
  const float* f; int fsq, a, hw;
  if (pp < 32000)       { f = f0; fsq = 6400; a = pp / 6400;          hw = pp % 6400; }
  else if (pp < 40000)  { int q = pp - 32000; f = f1; fsq = 1600; a = q / 1600; hw = q % 1600; }
  else                  { int q = pp - 40000; f = f2; fsq = 400;  a = q / 400;  hw = q % 400;  }
  const float* base = f + (size_t)(i * 425 + a * 85) * (size_t)fsq + hw;

  // build targets for this (image, prior)
  float ov = bo[i * M_PRIORS + pp];
  int   gi = bi[i * M_PRIORS + pp];
  float tclass = targets[gi * 6 + 1] + 1.0f;
  if (ov < IOU_TH) tclass = 0.0f;
  float posf = (tclass > 0.0f) ? 1.0f : 0.0f;
  int label = min(max((int)tclass - 1, 0), NCLS - 1);
  float4 pr = ((const float4*)priors)[pp];
  float bx = targets[gi*6+2], by = targets[gi*6+3], bw = targets[gi*6+4], bh = targets[gi*6+5];
  float tx  = (bx - pr.x) / pr.z, ty  = (by - pr.y) / pr.w;
  float twx = logf(bw / pr.z),    twy = logf(bh / pr.w);

  // channels 0..4: box + objectness (plane-strided, coalesced per lane)
  float px  = base[0];
  float py  = base[(size_t)fsq];
  float pw  = base[2 * (size_t)fsq];
  float ph  = base[3 * (size_t)fsq];
  float obj = base[4 * (size_t)fsq];
  float lbox = posf * (smooth_l1(px - tx) + smooth_l1(py - ty) +
                       smooth_l1(pw - twx) + smooth_l1(ph - twy));

  // channels 5..84: online logsumexp (single pass, exact)
  float mx = -__builtin_inff(), s = 0.0f, xl = 0.0f;
  const float* cb = base + 5 * (size_t)fsq;
  for (int cc = 0; cc < NCLS; ++cc) {
    float x = cb[(size_t)cc * (size_t)fsq];
    if (cc == label) xl = x;
    if (x > mx) { s = s * expf(mx - x) + 1.0f; mx = x; }
    else        { s += expf(x - mx); }
  }
  float ce  = (logf(s) + mx) - xl;
  float lcl = posf * ce;

  float l1e  = log1pf(expf(-fabsf(obj)));
  float bcev = fmaxf(obj, 0.0f) - obj * posf + l1e;               // BCE-with-logits
  float lcv  = (posf > 0.0f) ? 0.0f : (fmaxf(-obj, 0.0f) + l1e);  // -log_sigmoid, 0 at pos

  if (act) {
    lossc[i * M_PRIORS + p] = lcv;
    bcea [i * M_PRIORS + p] = bcev;
    pfa  [i * M_PRIORS + p] = posf;
  }
  // wave32 WMMA reductions (EXEC all-ones here; inactive lanes contribute 0)
  float rnp = wave_reduce_sum_wmma(act ? posf : 0.0f);
  float rlb = wave_reduce_sum_wmma(act ? lbox : 0.0f);
  float rlc = wave_reduce_sum_wmma(act ? lcl  : 0.0f);
  if ((threadIdx.x & 31) == 0) {
    atomicAdd(&scal[0], rnp);
    atomicAdd(&scal[1], rlb);
    atomicAdd(&scal[2], rlc);
  }
}

// ---------------- K3: exact per-image top-K threshold (64-bit radix select) -----
// key = (float bits of loss_c, ~index): distinct keys, larger value / smaller index
// ranks first => exact match of stable argsort-based `rank < num_neg`.
__global__ __launch_bounds__(256) void k_select(
    const float* __restrict__ lossc, const float* __restrict__ scal,
    unsigned long long* __restrict__ thr) {
  int i = blockIdx.x;
  float np = scal[0];
  int K = (int)fminf(np, (float)M_PRIORS - np);     // NEG_RATIO = 1
  if (K <= 0) { if (threadIdx.x == 0) thr[i] = ~0ULL; return; }

  __shared__ unsigned int hist[256];
  __shared__ unsigned long long pref_s;
  __shared__ int krem_s;
  if (threadIdx.x == 0) { pref_s = 0ULL; krem_s = K; }
  __syncthreads();

  for (int r = 0; r < 8; ++r) {
    hist[threadIdx.x] = 0;
    __syncthreads();
    unsigned long long prefix = pref_s;
    int shift = 56 - 8 * r;
    unsigned long long hmask = (r == 0) ? 0ULL : (~0ULL << (shift + 8));
    for (int p = threadIdx.x; p < M_PRIORS; p += 256) {
      float v = lossc[i * M_PRIORS + p];             // always >= 0 -> bits monotone
      unsigned long long key =
          ((unsigned long long)__float_as_uint(v) << 32) | (unsigned int)(~p);
      if ((key & hmask) == prefix)
        atomicAdd(&hist[(unsigned int)((key >> shift) & 0xFF)], 1u);
    }
    __syncthreads();
    if (threadIdx.x == 0) {
      int krem = krem_s, d = 0;
      for (int b = 255; b >= 0; --b) {
        int cnt = (int)hist[b];
        if (cnt >= krem) { d = b; break; }
        krem -= cnt;
      }
      pref_s = pref_s | ((unsigned long long)d << shift);
      krem_s = krem;
    }
    __syncthreads();
  }
  if (threadIdx.x == 0) thr[i] = pref_s;   // exactly K keys are >= thr
}

// ---------------- K4: objectness-loss sums over selected anchors ----------------
__global__ __launch_bounds__(256) void k_obj(
    const float* __restrict__ lossc, const float* __restrict__ bcea,
    const float* __restrict__ pfa, const unsigned long long* __restrict__ thr,
    float* __restrict__ scal) {
  int i = blockIdx.y;
  int p = blockIdx.x * 256 + threadIdx.x;
  bool act = p < M_PRIORS;
  float np = scal[0];
  int K = (int)fminf(np, (float)M_PRIORS - np);
  float selv = 0.0f, bcev = 0.0f;
  if (act) {
    float posf = pfa[i * M_PRIORS + p];
    float v = lossc[i * M_PRIORS + p];
    unsigned long long key =
        ((unsigned long long)__float_as_uint(v) << 32) | (unsigned int)(~p);
    bool neg = (K > 0) && (key >= thr[i]);
    if (posf > 0.0f || neg) { selv = 1.0f; bcev = bcea[i * M_PRIORS + p]; }
  }
  float rs = wave_reduce_sum_wmma(selv);
  float rb = wave_reduce_sum_wmma(bcev);
  if ((threadIdx.x & 31) == 0) {
    atomicAdd(&scal[3], rb);
    atomicAdd(&scal[4], rs);
  }
}

// ---------------- K5: assemble 4 scalar outputs ----------------
__global__ void k_final(const float* __restrict__ scal, float* __restrict__ out) {
  if (threadIdx.x == 0 && blockIdx.x == 0) {
    float np   = scal[0];
    float den2 = fmaxf(np * 2.0f, 1.0f);
    float lbox = scal[1] / den2;                    // (Sxy+Swh)/den2 == Sxy/den2+Swh/den2
    float lcls = scal[2] / fmaxf(np, 1.0f);
    float lobj = scal[3] / fmaxf(scal[4], 1.0f);
    out[0] = lbox + lcls + lobj;
    out[1] = lbox;
    out[2] = lobj;
    out[3] = lcls;
  }
}

// ---------------- host side ----------------
extern "C" void kernel_launch(void* const* d_in, const int* in_sizes, int n_in,
                              void* d_out, int out_size, void* d_ws, size_t ws_size,
                              hipStream_t stream) {
  const float* f0 = (const float*)d_in[0];
  const float* f1 = (const float*)d_in[1];
  const float* f2 = (const float*)d_in[2];
  const float* tg = (const float*)d_in[3];
  const float* pr = (const float*)d_in[4];

  // workspace layout (floats); total ~6.8 MB
  float* ws   = (float*)d_ws;
  float* scal = ws;                                        // [8] accumulators
  unsigned long long* thr = (unsigned long long*)(ws + 8); // [8] per-image thresholds
  float* bo = ws + 24;                                     // best overlap  [8*42000]
  int*   bi = (int*)(ws + 24 + (size_t)BATCH * M_PRIORS);  // best gt idx   [8*42000]
  int*   bp = (int*)(ws + 24 + 2 * (size_t)BATCH * M_PRIORS);          // [160]
  float* lc = ws + 24 + 2 * (size_t)BATCH * M_PRIORS + NT; // loss_c        [8*42000]
  float* bc = lc + (size_t)BATCH * M_PRIORS;               // bce           [8*42000]
  float* pf = bc + (size_t)BATCH * M_PRIORS;               // posf          [8*42000]

  dim3 gp(PB, BATCH);

  k_init<<<1, 32, 0, stream>>>(scal);
  k_best_per_prior<<<gp, 256, 0, stream>>>(tg, pr, bo, bi);
  k_best_per_target<<<NT, 256, 0, stream>>>(tg, pr, bp);
  k_scatter<<<1, 32, 0, stream>>>(bp, bo);
  k_main<<<gp, 256, 0, stream>>>(f0, f1, f2, tg, pr, bo, bi, lc, bc, pf, scal);
  k_select<<<BATCH, 256, 0, stream>>>(lc, scal, thr);
  k_obj<<<gp, 256, 0, stream>>>(lc, bc, pf, thr, scal);
  k_final<<<1, 32, 0, stream>>>(scal, (float*)d_out);
}
